// VectorQuantizer_41987600286127
// MI455X (gfx1250) — matrix operand order repaired and checked by reading, not compiled, not run on hardware
//
#include <hip/hip_runtime.h>
#include <hip/hip_bf16.h>
#include <math.h>

#define N_ROWS 131072
#define EMB    64
#define KB     2048
#define MUF    0.99f
#define CH     64          // k-rows per LDS chunk
#define NCHUNK (KB / CH)   // 32

typedef __attribute__((ext_vector_type(16))) __bf16 v16bf;
typedef __attribute__((ext_vector_type(8)))  float  v8f;

// ---- d_out flat layout (reference tuple order) ----
static constexpr size_t OFF_XL       = 0;                                   // int32 x 131072
static constexpr size_t OFF_XD       = OFF_XL + (size_t)N_ROWS;             // 131072*64
static constexpr size_t OFF_COMMIT   = OFF_XD + (size_t)N_ROWS * EMB;
static constexpr size_t OFF_FIT      = OFF_COMMIT + 1;
static constexpr size_t OFF_PRENORM  = OFF_FIT + 1;
static constexpr size_t OFF_ENTROPY  = OFF_PRENORM + 1;
static constexpr size_t OFF_USED     = OFF_ENTROPY + 1;                     // int32
static constexpr size_t OFF_USAGETOT = OFF_USED + 1;
static constexpr size_t OFF_DK       = OFF_USAGETOT + 1;
static constexpr size_t OFF_NEWK     = OFF_DK + 1;                          // 2048*64
static constexpr size_t OFF_NEWKSUM  = OFF_NEWK + (size_t)KB * EMB;         // 2048*64
static constexpr size_t OFF_NEWKELEM = OFF_NEWKSUM + (size_t)KB * EMB;      // 2048

// ---- workspace byte offsets ----
static constexpr size_t WSO_KBF   = 0;        // 2048*64 bf16   = 262144 B
static constexpr size_t WSO_KNORM = 262144;   // 2048 f32       =   8192 B
static constexpr size_t WSO_KSUM  = 270336;   // 2048*64 f32    = 524288 B
static constexpr size_t WSO_KELEM = 794624;   // 2048 f32       =   8192 B
static constexpr size_t WSO_SCAL  = 802816;   // 16 f32: 0=fit 1=sum_x 2=sum_x2 3=commit
static constexpr int    ZERO_FLOATS = KB * EMB + KB + 16;  // ksum_acc..scal contiguous

__device__ __forceinline__ __bf16 f2bf(float f) {
  unsigned u = __float_as_uint(f);
  u += 0x7FFFu + ((u >> 16) & 1u);          // round-to-nearest-even
  unsigned short s = (unsigned short)(u >> 16);
  __bf16 r; __builtin_memcpy(&r, &s, 2);
  return r;
}

// CDNA5 async global->LDS copy (ASYNCcnt-tracked DMA), 16B per lane.
__device__ __forceinline__ void async_copy_b128(unsigned lds_byte_addr, const void* gptr) {
  asm volatile("global_load_async_to_lds_b128 %0, %1, off"
               :: "v"(lds_byte_addr), "v"(gptr)
               : "memory");
}
__device__ __forceinline__ void wait_async0() {
  asm volatile("s_wait_asynccnt 0x0" ::: "memory");
}

__global__ void vq_zero(float* __restrict__ p, int n) {
  int i = blockIdx.x * blockDim.x + threadIdx.x;
  if (i < n) p[i] = 0.f;
}

// k (f32) -> bf16 copy (row-major) + per-row squared norms
__global__ void vq_prep_k(const float* __restrict__ k,
                          unsigned short* __restrict__ kbf,
                          float* __restrict__ knorm) {
  int j = blockIdx.x * blockDim.x + threadIdx.x;
  if (j >= KB) return;
  const float* kr = k + (size_t)j * EMB;
  unsigned short* o = kbf + (size_t)j * EMB;
  float ss = 0.f;
  #pragma unroll 8
  for (int t = 0; t < EMB; ++t) {
    float f = kr[t];
    ss += f * f;
    unsigned u = __float_as_uint(f);
    u += 0x7FFFu + ((u >> 16) & 1u);
    o[t] = (unsigned short)(u >> 16);
  }
  knorm[j] = ss;
}

// Main kernel: block = 8 waves x 32 x-rows = 256 rows. k is staged through a
// double-buffered LDS chunk (64 rows = 8KB) via async global->LDS DMA; every
// B slice read from LDS feeds 2 accumulator tiles (32 rows) -> 2x register
// reuse, 4 x v_wmma_f32_16x16x32_bf16 per 16-column tile (K=64).
__global__ void __launch_bounds__(256)
vq_argmin_wmma(const float* __restrict__ x,
               const unsigned short* __restrict__ kbf,
               const float* __restrict__ knorm,
               int* __restrict__ xl_out,
               float* __restrict__ scal) {
  __shared__ unsigned short kb[2][CH * EMB];   // 2 x 8KB

  const int tid  = threadIdx.x;
  const int lane = tid & 31;
  const int wave = tid >> 5;
  const int row0 = blockIdx.x * 256 + wave * 32;
  const int m    = lane & 15;   // A row / B,C column within tile
  const int half = lane >> 4;   // K-half carried by this lane

  // ---- A: two 16x64 bf16 row-tiles (rows row0+m, row0+16+m) ----
  v16bf a0[2], a1[2];
  float xnorm_full[2];
  #pragma unroll
  for (int r = 0; r < 2; ++r) {
    const float* xr = x + (size_t)(row0 + r * 16 + m) * EMB;
    float ss = 0.f;
    #pragma unroll
    for (int i = 0; i < 8; ++i) { float f = xr[half * 8 + i];      ss += f * f; a0[r][i]     = f2bf(f); }
    #pragma unroll
    for (int i = 0; i < 8; ++i) { float f = xr[16 + half * 8 + i]; ss += f * f; a0[r][8 + i] = f2bf(f); }
    #pragma unroll
    for (int i = 0; i < 8; ++i) { float f = xr[32 + half * 8 + i]; ss += f * f; a1[r][i]     = f2bf(f); }
    #pragma unroll
    for (int i = 0; i < 8; ++i) { float f = xr[48 + half * 8 + i]; ss += f * f; a1[r][8 + i] = f2bf(f); }
    xnorm_full[r] = ss + __shfl_xor(ss, 16);   // partner lane has other 32 elems
  }

  // running min of (||k_j||^2 - 2 x.k_j); ||x||^2 is j-invariant -> added at end
  float minv[2][8]; int mini[2][8];
  #pragma unroll
  for (int r = 0; r < 2; ++r)
    #pragma unroll
    for (int i = 0; i < 8; ++i) { minv[r][i] = 3.4e38f; mini[r][i] = 0; }

  const unsigned kb_base = (unsigned)(uintptr_t)&kb[0][0];

  // issue chunk 0
  {
    const char* src = (const char*)kbf + (size_t)tid * 32;
    unsigned    dst = kb_base + (unsigned)tid * 32;
    async_copy_b128(dst,      src);
    async_copy_b128(dst + 16, src + 16);
  }

  for (int c = 0; c < NCHUNK; ++c) {
    wait_async0();        // this wave's copies for chunk c complete
    __syncthreads();      // all waves' copies visible; buffers free for reuse

    if (c + 1 < NCHUNK) { // prefetch next chunk into the other buffer (overlaps compute)
      const char* src = (const char*)kbf + (size_t)(c + 1) * CH * EMB * 2 + (size_t)tid * 32;
      unsigned    dst = kb_base + (unsigned)(((c + 1) & 1) * (CH * EMB * 2)) + (unsigned)tid * 32;
      async_copy_b128(dst,      src);
      async_copy_b128(dst + 16, src + 16);
    }

    const unsigned short* kc  = &kb[c & 1][0];
    const float*          knc = knorm + c * CH;

    #pragma unroll
    for (int jj = 0; jj < CH / 16; ++jj) {
      const int j = c * CH + jj * 16 + m;                  // this lane's column
      const unsigned short* krow = kc + (size_t)(jj * 16 + m) * EMB;
      // B 32x16 layout: lane n = col; lanes 0-15 carry K 0..7,16..23, lanes 16-31 K 8..15,24..31
      union { uint4 u[2]; v16bf v; } b0u, b1u;
      const uint4* p0 = (const uint4*)(krow + half * 8);        // ds_load_b128 x2
      b0u.u[0] = p0[0]; b0u.u[1] = p0[2];
      const uint4* p1 = (const uint4*)(krow + 32 + half * 8);
      b1u.u[0] = p1[0]; b1u.u[1] = p1[2];
      const float kn = knc[jj * 16 + m];

      #pragma unroll
      for (int r = 0; r < 2; ++r) {
        v8f cc = {};
        cc = __builtin_amdgcn_wmma_f32_16x16x32_bf16(false, a0[r], false, b0u.v, (short)0, cc, false, false);
        cc = __builtin_amdgcn_wmma_f32_16x16x32_bf16(false, a1[r], false, b1u.v, (short)0, cc, false, false);
        #pragma unroll
        for (int i = 0; i < 8; ++i) {
          float d = fmaf(-2.f, cc[i], kn);                 // single v_fma per element
          if (d < minv[r][i]) { minv[r][i] = d; mini[r][i] = j; }
        }
      }
    }
  }

  // reduce (min, idx) across 16 columns; masks 8..1 stay within 16-lane halves
  const int mbase = (lane < 16) ? 0 : 8;   // C layout: low lanes M=0..7, high M=8..15
  #pragma unroll
  for (int r = 0; r < 2; ++r) {
    #pragma unroll
    for (int i = 0; i < 8; ++i) {
      #pragma unroll
      for (int s = 8; s >= 1; s >>= 1) {
        float ov = __shfl_xor(minv[r][i], s);
        int   oi = __shfl_xor(mini[r][i], s);
        if (ov < minv[r][i] || (ov == minv[r][i] && oi < mini[r][i])) {
          minv[r][i] = ov; mini[r][i] = oi;
        }
      }
    }
  }

  if ((lane & 15) == 0) {   // lane 0 -> rows mbase 0..7, lane 16 -> rows 8..15
    float fs = 0.f;
    #pragma unroll
    for (int r = 0; r < 2; ++r) {
      #pragma unroll
      for (int i = 0; i < 8; ++i) {
        const int rowi = mbase + i;
        xl_out[row0 + r * 16 + rowi] = mini[r][i];
        fs += minv[r][i] + __shfl(xnorm_full[r], rowi);    // re-add ||x||^2 for fit
      }
    }
    atomicAdd(&scal[0], fs);
  }
}

// x_d = k[x_l] gather + segment sums + commit/prenorm partials
__global__ void __launch_bounds__(256)
vq_gather_segsum(const float* __restrict__ x, const float* __restrict__ k,
                 const int* __restrict__ xl, float* __restrict__ xd,
                 float* __restrict__ ksum, float* __restrict__ kelem,
                 float* __restrict__ scal) {
  const int i = blockIdx.x * blockDim.x + threadIdx.x;
  const int l = xl[i];
  const float4* xr = (const float4*)(x + (size_t)i * EMB);
  const float4* kr = (const float4*)(k + (size_t)l * EMB);
  float4* xo = (float4*)(xd + (size_t)i * EMB);
  float* ks = ksum + (size_t)l * EMB;
  float commit = 0.f, sx = 0.f, sx2 = 0.f;
  #pragma unroll 4
  for (int t = 0; t < 16; ++t) {
    float4 xv = xr[t], kv = kr[t];
    xo[t] = kv;
    float d;
    d = kv.x - xv.x; commit += d * d; sx += xv.x; sx2 += xv.x * xv.x; atomicAdd(&ks[t * 4 + 0], xv.x);
    d = kv.y - xv.y; commit += d * d; sx += xv.y; sx2 += xv.y * xv.y; atomicAdd(&ks[t * 4 + 1], xv.y);
    d = kv.z - xv.z; commit += d * d; sx += xv.z; sx2 += xv.z * xv.z; atomicAdd(&ks[t * 4 + 2], xv.z);
    d = kv.w - xv.w; commit += d * d; sx += xv.w; sx2 += xv.w * xv.w; atomicAdd(&ks[t * 4 + 3], xv.w);
  }
  atomicAdd(&kelem[l], 1.f);
  #pragma unroll
  for (int s = 16; s > 0; s >>= 1) {
    commit += __shfl_xor(commit, s);
    sx     += __shfl_xor(sx, s);
    sx2    += __shfl_xor(sx2, s);
  }
  if ((threadIdx.x & 31) == 0) {
    atomicAdd(&scal[3], commit);
    atomicAdd(&scal[1], sx);
    atomicAdd(&scal[2], sx2);
  }
}

__device__ __forceinline__ float block_reduce(float v, float* red) {
  int t = threadIdx.x;
  red[t] = v; __syncthreads();
  for (int s = 128; s > 0; s >>= 1) { if (t < s) red[t] += red[t + s]; __syncthreads(); }
  float r = red[0]; __syncthreads();
  return r;
}

// single-block epilogue: EMA update, entropy, dk, scalar outputs
__global__ void __launch_bounds__(256)
vq_finalize(const float* __restrict__ x, const float* __restrict__ k,
            const float* __restrict__ ksum_in, const float* __restrict__ kelem_in,
            const float* __restrict__ ksum_acc, const float* __restrict__ kelem_acc,
            const float* __restrict__ scal, float* __restrict__ out) {
  __shared__ float red[256];
  const int t = threadIdx.x;

  float tot = 0.f;
  for (int j = t; j < KB; j += 256) tot += kelem_acc[j];
  const float elem_total = block_reduce(tot, red);

  float* new_k      = out + OFF_NEWK;
  float* new_k_sum  = out + OFF_NEWKSUM;
  float* new_k_elem = out + OFF_NEWKELEM;

  float ent = 0.f, used = 0.f, usage_tot = 0.f, dkacc = 0.f;
  for (int j = t; j < KB; j += 256) {
    float e   = kelem_acc[j];
    float nke = MUF * kelem_in[j] + (1.f - MUF) * e;
    new_k_elem[j] = nke;
    float usage = (nke >= 1.f) ? 1.f : 0.f;
    usage_tot += usage;
    used += (e >= 1.f) ? 1.f : 0.f;
    float p = e / elem_total;
    ent -= p * logf(p + 1e-8f);
    // deterministic stand-in for x[perm[:K]] (host RNG not reproducible here)
    const float* xrand = x + (size_t)j * 64 * EMB;
    for (int d = 0; d < EMB; ++d) {
      float nks = MUF * ksum_in[j * EMB + d] + (1.f - MUF) * ksum_acc[j * EMB + d];
      new_k_sum[j * EMB + d] = nks;
      float nk = (usage > 0.5f) ? (nks / nke) : xrand[d];
      new_k[j * EMB + d] = nk;
      float dd = nk - k[j * EMB + d];
      dkacc += dd * dd;
    }
  }
  const float entropy = block_reduce(ent, red);
  const float usedc   = block_reduce(used, red);
  const float usaget  = block_reduce(usage_tot, red);
  const float dks     = block_reduce(dkacc, red);

  if (t == 0) {
    const float sizeX = (float)N_ROWS * (float)EMB;
    float fit    = scal[0] / (float)N_ROWS;
    float commit = scal[3] / sizeX;
    float mean   = scal[1] / sizeX;
    float pren   = sqrtf(fmaxf(scal[2] - sizeX * mean * mean, 0.f)) / sqrtf(sizeX);
    float dk     = sqrtf(dks) / sqrtf((float)KB * (float)EMB);
    out[OFF_COMMIT]   = commit;
    out[OFF_FIT]      = fit;
    out[OFF_PRENORM]  = pren;
    out[OFF_ENTROPY]  = entropy;
    ((int*)out)[OFF_USED] = (int)(usedc + 0.5f);
    out[OFF_USAGETOT] = usaget;
    out[OFF_DK]       = dk;
  }
}

extern "C" void kernel_launch(void* const* d_in, const int* in_sizes, int n_in,
                              void* d_out, int out_size, void* d_ws, size_t ws_size,
                              hipStream_t stream) {
  const float* x      = (const float*)d_in[0];
  const float* k      = (const float*)d_in[1];
  const float* k_sum  = (const float*)d_in[2];
  const float* k_elem = (const float*)d_in[3];
  float* out = (float*)d_out;
  char*  ws  = (char*)d_ws;

  unsigned short* kbf = (unsigned short*)(ws + WSO_KBF);
  float* knorm     = (float*)(ws + WSO_KNORM);
  float* ksum_acc  = (float*)(ws + WSO_KSUM);
  float* kelem_acc = (float*)(ws + WSO_KELEM);
  float* scal      = (float*)(ws + WSO_SCAL);

  vq_zero<<<(ZERO_FLOATS + 255) / 256, 256, 0, stream>>>(ksum_acc, ZERO_FLOATS);
  vq_prep_k<<<(KB + 255) / 256, 256, 0, stream>>>(k, kbf, knorm);
  vq_argmin_wmma<<<N_ROWS / 256, 256, 0, stream>>>(x, kbf, knorm, (int*)d_out, scal);
  vq_gather_segsum<<<N_ROWS / 256, 256, 0, stream>>>(x, k, (const int*)d_out,
                                                     out + OFF_XD, ksum_acc, kelem_acc, scal);
  vq_finalize<<<1, 256, 0, stream>>>(x, k, k_sum, k_elem, ksum_acc, kelem_acc, scal, out);
}
